// VanillaRNN_82781199663717
// MI455X (gfx1250) — matrix operand order repaired and checked by reading, not compiled, not run on hardware
//
#include <hip/hip_runtime.h>
#include <hip/hip_bf16.h>
#include <math.h>

// ---------------------------------------------------------------------------
// VanillaRNN on gfx1250:  h_{t+1} = tanh([h_t | x_t] @ [Whh ; Whx] + bh)
// fused-K BF16 WMMA, register-resident weights, persistent kernel with
// device-wide barrier; then out = softmax(h_T @ Wph + bo).
// ---------------------------------------------------------------------------

typedef __bf16 bf16_t;
typedef bf16_t v8bf  __attribute__((ext_vector_type(8)));
typedef bf16_t v16bf __attribute__((ext_vector_type(16)));
typedef float  v8f   __attribute__((ext_vector_type(8)));

constexpr int BATCH = 256;
constexpr int TSEQ  = 512;
constexpr int INDIM = 64;
constexpr int HDIM  = 512;
constexpr int KTOT  = HDIM + INDIM;   // 576 = 18 chunks of 32
constexpr int NBLK  = 64;             // persistent workgroups (8 waves each)

// round-to-nearest-even f32 -> bf16
static __device__ __forceinline__ bf16_t f2bf(float f) {
  unsigned u = __builtin_bit_cast(unsigned, f);
  u += 0x7FFFu + ((u >> 16) & 1u);
  unsigned short s = (unsigned short)(u >> 16);
  return __builtin_bit_cast(bf16_t, s);
}

// A-fragment (16x32 bf16): lane holds K=kb..kb+7 and K=kb+16..kb+23
// as two contiguous 16-byte chunks from a row-major row.
static __device__ __forceinline__ v16bf load_a_frag(const bf16_t* p) {
  v8bf lo = *(const v8bf*)p;
  v8bf hi = *(const v8bf*)(p + 16);
  return __builtin_shufflevector(lo, hi,
      0, 1, 2, 3, 4, 5, 6, 7, 8, 9, 10, 11, 12, 13, 14, 15);
}

// --- prep: zero h0 and barrier state ---------------------------------------
__global__ void init_state(unsigned* __restrict__ hA_u, int* __restrict__ bar) {
  int i = blockIdx.x * blockDim.x + threadIdx.x;   // 65536 dwords = 256x512 bf16
  hA_u[i] = 0u;
  if (i == 0) { bar[0] = 0; bar[1] = 0; }
}

// --- prep: x f32 -> bf16 (same (B,T,I) row-major layout) -------------------
__global__ void cvt_x(const float* __restrict__ x, bf16_t* __restrict__ xbf) {
  int i = blockIdx.x * blockDim.x + threadIdx.x;
  xbf[i] = f2bf(x[i]);
}

// --- prep: build swizzled B fragments for fused weights [Whh ; Whx] --------
// B-fragment lane order (32x16 bf16): lane<16 -> N=lane, K=0..7,16..23;
// lane>=16 -> N=lane-16, K=8..15,24..31. Stored so the main kernel reads 32
// contiguous bytes per lane per chunk.
__global__ void pack_weights(const float* __restrict__ Whx,
                             const float* __restrict__ Whh,
                             bf16_t* __restrict__ Bbuf) {
  int b      = blockIdx.x;           // 0..575 = n_tile*18 + kc
  int n_tile = b / 18;
  int kc     = b % 18;
  int lane   = threadIdx.x >> 4;     // 0..31
  int j      = threadIdx.x & 15;     // element within lane
  int K = kc * 32 + (lane >> 4) * 8 + (j & 7) + (j >> 3) * 16;
  int N = n_tile * 16 + (lane & 15);
  float w = (K < HDIM) ? Whh[(size_t)K * HDIM + N]
                       : Whx[(size_t)(K - HDIM) * HDIM + N];
  Bbuf[((size_t)b * 32 + lane) * 16 + j] = f2bf(w);
}

// --- main persistent recurrence kernel -------------------------------------
__global__ __launch_bounds__(256, 1)
void rnn_persistent(const bf16_t* __restrict__ xbf,
                    const bf16_t* __restrict__ Bbuf,
                    const float*  __restrict__ bh,
                    bf16_t* __restrict__ hA,
                    bf16_t* __restrict__ hB,
                    int* __restrict__ bar) {
  const int lane = threadIdx.x & 31;
  const int wave = threadIdx.x >> 5;
  const int tile = blockIdx.x * 8 + wave;   // 0..511 unique 16x16 output tiles
  const int m_tile = tile & 15;             // 16 row tiles  (BATCH/16)
  const int n_tile = tile >> 4;             // 32 col tiles  (HDIM/16)

  const int row = m_tile * 16 + (lane & 15);   // A-fragment source row
  const int kb  = (lane >> 4) * 8;             // A-fragment K sub-offset
  const int col = n_tile * 16 + (lane & 15);   // C/D column for this lane

  // Load this wave's 18 B fragments ONCE; they live in VGPRs for all 512 steps.
  v16bf bw[18];
  {
    const bf16_t* bp = Bbuf + ((size_t)(n_tile * 18) * 32 + lane) * 16;
#pragma unroll
    for (int kc = 0; kc < 18; ++kc)
      bw[kc] = *(const v16bf*)(bp + (size_t)kc * 32 * 16);
  }
  const float bias = bh[col];

  bf16_t* cur = hA;
  bf16_t* nxt = hB;
  const bf16_t* xrow = xbf + (size_t)row * (TSEQ * INDIM);

  for (int t = 0; t < TSEQ; ++t) {
    v8f acc = {};

    // K = 0..511 : h_t @ Whh
    const bf16_t* hp = cur + (size_t)row * HDIM + kb;
#pragma unroll
    for (int kc = 0; kc < 16; ++kc)
      acc = __builtin_amdgcn_wmma_f32_16x16x32_bf16(
          false, load_a_frag(hp + kc * 32), false, bw[kc],
          (short)0, acc, false, false);

    // K = 512..575 : x_t @ Whx (fused tail)
    const bf16_t* xp = xrow + (size_t)t * INDIM + kb;
    acc = __builtin_amdgcn_wmma_f32_16x16x32_bf16(
        false, load_a_frag(xp), false, bw[16], (short)0, acc, false, false);
    acc = __builtin_amdgcn_wmma_f32_16x16x32_bf16(
        false, load_a_frag(xp + 32), false, bw[17], (short)0, acc, false, false);

    if (t + 1 < TSEQ)   // pull next timestep's x slice toward the WGP
      __builtin_prefetch(xrow + (size_t)(t + 1) * INDIM, 0, 3);

    // D layout: lane -> column `col`, VGPR v -> row (lane>>4)*8 + v
    bf16_t* op = nxt + (size_t)(m_tile * 16 + (lane >> 4) * 8) * HDIM + col;
#pragma unroll
    for (int v = 0; v < 8; ++v)
      op[(size_t)v * HDIM] = f2bf(tanhf(acc[v] + bias));

    // ---- device-wide barrier (all 64 persistent WGs) ----
    __threadfence();          // release h_next to agent scope
    __syncthreads();
    if (threadIdx.x == 0) {
      int gen  = __hip_atomic_load(&bar[1], __ATOMIC_ACQUIRE,
                                   __HIP_MEMORY_SCOPE_AGENT);
      int prev = __hip_atomic_fetch_add(&bar[0], 1, __ATOMIC_ACQ_REL,
                                        __HIP_MEMORY_SCOPE_AGENT);
      if (prev == NBLK - 1) {
        __hip_atomic_store(&bar[0], 0, __ATOMIC_RELAXED,
                           __HIP_MEMORY_SCOPE_AGENT);
        __hip_atomic_fetch_add(&bar[1], 1, __ATOMIC_RELEASE,
                               __HIP_MEMORY_SCOPE_AGENT);
      } else {
        while (__hip_atomic_load(&bar[1], __ATOMIC_ACQUIRE,
                                 __HIP_MEMORY_SCOPE_AGENT) == gen)
          __builtin_amdgcn_s_sleep(1);
      }
    }
    __syncthreads();
    __threadfence();          // acquire: see all other WGs' h_next

    bf16_t* tmp = cur; cur = nxt; nxt = tmp;
  }
  // TSEQ even -> final hidden state ends up in hA
}

// --- epilogue: out = softmax(h_T @ Wph + bo), one block per batch row ------
__global__ __launch_bounds__(256)
void out_softmax(const bf16_t* __restrict__ h,
                 const float* __restrict__ Wph,
                 const float* __restrict__ bo,
                 float* __restrict__ out) {
  const int b   = blockIdx.x;
  const int tid = threadIdx.x;            // 256 threads, 2 outputs each
  __shared__ float hrow[HDIM];
  __shared__ float red[256];

  hrow[tid]       = (float)h[(size_t)b * HDIM + tid];
  hrow[tid + 256] = (float)h[(size_t)b * HDIM + tid + 256];
  __syncthreads();

  float a0 = bo[tid], a1 = bo[tid + 256];
  for (int k = 0; k < HDIM; ++k) {
    float hv = hrow[k];
    a0 += hv * Wph[(size_t)k * HDIM + tid];
    a1 += hv * Wph[(size_t)k * HDIM + tid + 256];
  }

  red[tid] = fmaxf(a0, a1);
  __syncthreads();
  for (int s = 128; s > 0; s >>= 1) {
    if (tid < s) red[tid] = fmaxf(red[tid], red[tid + s]);
    __syncthreads();
  }
  float mx = red[0];
  __syncthreads();

  float e0 = expf(a0 - mx), e1 = expf(a1 - mx);
  red[tid] = e0 + e1;
  __syncthreads();
  for (int s = 128; s > 0; s >>= 1) {
    if (tid < s) red[tid] += red[tid + s];
    __syncthreads();
  }
  float inv = 1.0f / red[0];
  out[(size_t)b * HDIM + tid]       = e0 * inv;
  out[(size_t)b * HDIM + tid + 256] = e1 * inv;
}

// ---------------------------------------------------------------------------
extern "C" void kernel_launch(void* const* d_in, const int* in_sizes, int n_in,
                              void* d_out, int out_size, void* d_ws,
                              size_t ws_size, hipStream_t stream) {
  (void)in_sizes; (void)n_in; (void)out_size; (void)ws_size;

  const float* x   = (const float*)d_in[0];
  const float* Whx = (const float*)d_in[1];
  const float* Whh = (const float*)d_in[2];
  const float* bh  = (const float*)d_in[3];
  const float* Wph = (const float*)d_in[4];
  const float* bo  = (const float*)d_in[5];
  float* out = (float*)d_out;

  // workspace layout (~18 MB total)
  char* ws = (char*)d_ws;
  constexpr size_t HBYTES = (size_t)BATCH * HDIM * sizeof(bf16_t);      // 256 KB
  constexpr size_t XBYTES = (size_t)BATCH * TSEQ * INDIM * sizeof(bf16_t); // 16 MB
  int*    bar  = (int*)ws;                                   // [count, gen]
  bf16_t* hA   = (bf16_t*)(ws + 4096);
  bf16_t* hB   = (bf16_t*)(ws + 4096 + HBYTES);
  bf16_t* xbf  = (bf16_t*)(ws + 4096 + 2 * HBYTES);
  bf16_t* Bbuf = (bf16_t*)(ws + 4096 + 2 * HBYTES + XBYTES); // 576 KB

  init_state<<<(BATCH * HDIM / 2) / 256, 256, 0, stream>>>((unsigned*)hA, bar);
  cvt_x<<<(BATCH * TSEQ * INDIM) / 256, 256, 0, stream>>>(x, xbf);
  pack_weights<<<(HDIM / 16) * (KTOT / 32), 512, 0, stream>>>(Whx, Whh, Bbuf);
  rnn_persistent<<<NBLK, 256, 0, stream>>>(xbf, Bbuf, bh, hA, hB, bar);
  out_softmax<<<BATCH, 256, 0, stream>>>(hA, Wph, bo, out);
}
